// Decoder_55173149884490
// MI455X (gfx1250) — compile-verified, matrix operand order checked by reading, not run on hardware
//
#include <hip/hip_runtime.h>
#include <hip/hip_bf16.h>

// Problem constants
#define BB 32
#define SS 2048
#define HH 256
#define EE 128
#define VV 50000
#define OOVV 200
#define H2H 512
#define VE (VV + OOVV)   // 50200
#define NCTX 16          // context partial chunks per batch row
#define NST 8            // row-stat partial chunks per batch row

// d_out layout (floats), reference tuple order
#define O_FINAL 0
#define O_H   (BB * VE)                 // 1,606,400
#define O_C   (O_H + BB * HH)
#define O_CTX (O_C + BB * HH)
#define O_ATTN (O_CTX + BB * H2H)
#define O_PGEN (O_ATTN + BB * SS)
#define O_COV  (O_PGEN + BB)

// d_ws layout (floats)
#define WS_X    0                         // x_comb: B*E
#define WS_DF   (WS_X + BB * EE)          // dec_feat: B*H2
#define WS_SC   (WS_DF + BB * H2H)        // scores: B*S
#define WS_O1   (WS_SC + BB * SS)         // out1: B*H
#define WS_MAX  (WS_O1 + BB * HH)         // rowmax: B
#define WS_SCL  (WS_MAX + BB)             // pgen/rowsum: B
#define WS_PM   (WS_SCL + BB)             // partial max: B*NST
#define WS_PS   (WS_PM + BB * NST)        // partial sum: B*NST
#define WS_CTXP (WS_PS + BB * NST)        // partial context: B*NCTX*H2
#define WS_LOG  (WS_CTXP + BB * NCTX * H2H)  // logits: B*V  (total ~7.9 MB)

typedef __attribute__((ext_vector_type(16))) _Float16 v16h;
typedef __attribute__((ext_vector_type(8)))  _Float16 v8h;
typedef __attribute__((ext_vector_type(8)))  float    v8f;

__device__ __forceinline__ float sigmf(float x) { return 1.f / (1.f + __expf(-x)); }

// ---------------------------------------------------------------------------
// K1: embed gather + [prev_ctx,x]@W_comb.T + LSTM step + dec_feat = ds@W_dp.T
// one block per batch row, everything LDS resident
// ---------------------------------------------------------------------------
__global__ __launch_bounds__(256) void k1_embed_lstm(
    const int* __restrict__ inp, const float* __restrict__ h0,
    const float* __restrict__ c0, const float* __restrict__ prev_ctx,
    const float* __restrict__ emb, const float* __restrict__ W_comb,
    const float* __restrict__ b_comb, const float* __restrict__ W_ih,
    const float* __restrict__ W_hh, const float* __restrict__ b_ih,
    const float* __restrict__ b_hh, const float* __restrict__ W_dp,
    const float* __restrict__ b_dp,
    float* __restrict__ x_comb_ws, float* __restrict__ dec_feat_ws,
    float* __restrict__ out_h, float* __restrict__ out_c)
{
  __shared__ float s_in[H2H + EE];   // 640: [prev_context, emb]
  __shared__ float s_x[EE];
  __shared__ float s_h0[HH];
  __shared__ float s_g[4 * HH];
  __shared__ float s_ds[H2H];
  const int b = blockIdx.x, t = threadIdx.x;

  s_in[t]       = prev_ctx[b * H2H + t];
  s_in[t + 256] = prev_ctx[b * H2H + t + 256];
  const int tok = inp[b];
  if (t < EE) s_in[H2H + t] = emb[tok * EE + t];
  if (t < HH) s_h0[t] = h0[b * HH + t];
  __syncthreads();

  if (t < EE) {
    float acc = b_comb[t];
    const float* w = W_comb + t * (H2H + EE);
    #pragma unroll 8
    for (int k = 0; k < H2H + EE; ++k) acc += s_in[k] * w[k];
    s_x[t] = acc;
    x_comb_ws[b * EE + t] = acc;
  }
  __syncthreads();

  #pragma unroll
  for (int q = 0; q < 4; ++q) {
    const int j = q * 256 + t;
    float acc = b_ih[j] + b_hh[j];
    const float* wi = W_ih + j * EE;
    #pragma unroll 8
    for (int k = 0; k < EE; ++k) acc += s_x[k] * wi[k];
    const float* wh = W_hh + j * HH;
    #pragma unroll 8
    for (int k = 0; k < HH; ++k) acc += s_h0[k] * wh[k];
    s_g[j] = acc;
  }
  __syncthreads();

  if (t < HH) {  // torch gate order i,f,g,o
    const float ig = sigmf(s_g[t]);
    const float fg = sigmf(s_g[HH + t]);
    const float gg = tanhf(s_g[2 * HH + t]);
    const float og = sigmf(s_g[3 * HH + t]);
    const float c = fg * c0[b * HH + t] + ig * gg;
    const float h = og * tanhf(c);
    out_h[b * HH + t] = h;
    out_c[b * HH + t] = c;
    s_ds[t] = h; s_ds[HH + t] = c;
  }
  __syncthreads();

  #pragma unroll
  for (int q = 0; q < 2; ++q) {
    const int j = q * 256 + t;
    float acc = b_dp[j];
    const float* w = W_dp + j * H2H;
    #pragma unroll 8
    for (int k = 0; k < H2H; ++k) acc += s_ds[k] * w[k];
    dec_feat_ws[b * H2H + j] = acc;
  }
}

// ---------------------------------------------------------------------------
// K2: word-attention scores, streams encoder_feature (134 MB) once.
// 8 waves/block, each wave owns one s at a time; per-lane float4 chunks,
// wave32 shfl_xor reduction. 1024 blocks -> plenty of HBM concurrency.
// ---------------------------------------------------------------------------
__global__ __launch_bounds__(256) void k2_scores(
    const float* __restrict__ enc_feat, const float* __restrict__ dec_feat,
    const float* __restrict__ cov, const float* __restrict__ W_c,
    const float* __restrict__ v_attn, float* __restrict__ scores)
{
  const int b = blockIdx.x;
  const int wave = threadIdx.x >> 5, lane = threadIdx.x & 31;
  float4 vq[4], wq[4], dq[4];
  #pragma unroll
  for (int i = 0; i < 4; ++i) {
    const int d = i * 128 + lane * 4;
    vq[i] = *(const float4*)(v_attn + d);
    wq[i] = *(const float4*)(W_c + d);
    dq[i] = *(const float4*)(dec_feat + b * H2H + d);
  }
  const int s0 = blockIdx.y * 64 + wave * 8;
  for (int si = 0; si < 8; ++si) {
    const int s = s0 + si;
    const float* ef = enc_feat + ((size_t)(b * SS + s)) * H2H;
    const float cv = cov[b * SS + s];
    float acc = 0.f;
    #pragma unroll
    for (int i = 0; i < 4; ++i) {
      const float4 e = *(const float4*)(ef + i * 128 + lane * 4);
      acc += vq[i].x * tanhf(e.x + dq[i].x + cv * wq[i].x);
      acc += vq[i].y * tanhf(e.y + dq[i].y + cv * wq[i].y);
      acc += vq[i].z * tanhf(e.z + dq[i].z + cv * wq[i].z);
      acc += vq[i].w * tanhf(e.w + dq[i].w + cv * wq[i].w);
    }
    #pragma unroll
    for (int off = 16; off > 0; off >>= 1) acc += __shfl_xor(acc, off, 32);
    if (lane == 0) scores[b * SS + s] = acc;
  }
}

// ---------------------------------------------------------------------------
// K3a: masked softmax over S + coverage_next (one block of 512 per row).
// exp values stay in registers; only the 512-wide reduction uses LDS.
// ---------------------------------------------------------------------------
__global__ __launch_bounds__(512) void k3a_softmax(
    const float* __restrict__ scores, const float* __restrict__ mask,
    const float* __restrict__ cov_in,
    float* __restrict__ attn_out, float* __restrict__ cov_out)
{
  __shared__ float s_red[512];
  const int b = blockIdx.x, t = threadIdx.x;

  float loc[4];
  float m = -INFINITY;
  #pragma unroll
  for (int q = 0; q < 4; ++q) { loc[q] = scores[b * SS + q * 512 + t]; m = fmaxf(m, loc[q]); }
  s_red[t] = m; __syncthreads();
  for (int off = 256; off > 0; off >>= 1) {
    if (t < off) s_red[t] = fmaxf(s_red[t], s_red[t + off]);
    __syncthreads();
  }
  m = s_red[0]; __syncthreads();

  float sum = 0.f;
  #pragma unroll
  for (int q = 0; q < 4; ++q) {
    const int s = q * 512 + t;
    loc[q] = __expf(loc[q] - m) * mask[b * SS + s];
    sum += loc[q];
  }
  s_red[t] = sum; __syncthreads();
  for (int off = 256; off > 0; off >>= 1) {
    if (t < off) s_red[t] += s_red[t + off];
    __syncthreads();
  }
  const float inv = 1.f / s_red[0];

  #pragma unroll
  for (int q = 0; q < 4; ++q) {
    const int s = q * 512 + t;
    const float a = loc[q] * inv;
    attn_out[b * SS + s] = a;
    cov_out[b * SS + s] = cov_in[b * SS + s] + a;
  }
}

// ---------------------------------------------------------------------------
// K3b: partial context: grid (B, NCTX); each block streams 128 rows of
// encoder_outputs (256 KB) coalesced -> 512 blocks in flight for the 134 MB
// stream. Deterministic: partials combined in fixed order by k3c.
// ---------------------------------------------------------------------------
__global__ __launch_bounds__(512) void k3b_ctx_partial(
    const float* __restrict__ attn, const float* __restrict__ enc_out,
    float* __restrict__ ctx_partial)
{
  __shared__ float s_a[SS / NCTX];   // 128 attention weights
  const int b = blockIdx.x, chunk = blockIdx.y, t = threadIdx.x;
  const int s0 = chunk * (SS / NCTX);
  if (t < SS / NCTX) s_a[t] = attn[b * SS + s0 + t];
  __syncthreads();

  float acc = 0.f;
  const float* eo = enc_out + ((size_t)b * SS + s0) * H2H + t;
  #pragma unroll 8
  for (int s = 0; s < SS / NCTX; ++s) acc += s_a[s] * eo[(size_t)s * H2H];
  ctx_partial[(size_t)(b * NCTX + chunk) * H2H + t] = acc;
}

// K3c: deterministic fixed-order combine of context partials
__global__ __launch_bounds__(512) void k3c_ctx_combine(
    const float* __restrict__ ctx_partial, float* __restrict__ ctx_out)
{
  const int b = blockIdx.x, t = threadIdx.x;
  float acc = 0.f;
  #pragma unroll
  for (int c = 0; c < NCTX; ++c)
    acc += ctx_partial[(size_t)(b * NCTX + c) * H2H + t];
  ctx_out[b * H2H + t] = acc;
}

// ---------------------------------------------------------------------------
// K4: p_gen gate + out1 = [h_new, context] @ W_out1.T + b_out1
// ---------------------------------------------------------------------------
__global__ __launch_bounds__(128) void k4_pgen_out1(
    const float* __restrict__ ctx, const float* __restrict__ hN,
    const float* __restrict__ cN, const float* __restrict__ x_comb,
    const float* __restrict__ W_pgen, const float* __restrict__ b_pgen,
    const float* __restrict__ W_out1, const float* __restrict__ b_out1,
    float* __restrict__ pgen_out, float* __restrict__ out1_ws)
{
  __shared__ float s_v[1152];  // [context(512), h_new(256), c_new(256), x(128)]
  __shared__ float s_red[128];
  const int b = blockIdx.x, t = threadIdx.x;
  #pragma unroll
  for (int q = 0; q < 4; ++q) s_v[q * 128 + t] = ctx[b * H2H + q * 128 + t];
  #pragma unroll
  for (int q = 0; q < 2; ++q) s_v[512 + q * 128 + t] = hN[b * HH + q * 128 + t];
  #pragma unroll
  for (int q = 0; q < 2; ++q) s_v[768 + q * 128 + t] = cN[b * HH + q * 128 + t];
  s_v[1024 + t] = x_comb[b * EE + t];
  __syncthreads();

  float acc = 0.f;
  #pragma unroll
  for (int q = 0; q < 9; ++q) acc += s_v[q * 128 + t] * W_pgen[q * 128 + t];
  s_red[t] = acc; __syncthreads();
  for (int off = 64; off > 0; off >>= 1) {
    if (t < off) s_red[t] += s_red[t + off];
    __syncthreads();
  }
  if (t == 0) pgen_out[b] = sigmf(s_red[0] + b_pgen[0]);

  #pragma unroll
  for (int q = 0; q < 2; ++q) {
    const int j = q * 128 + t;
    float a2 = b_out1[j];
    const float* w = W_out1 + j * 768;
    #pragma unroll 8
    for (int k = 0; k < 256; ++k) a2 += s_v[512 + k] * w[k];        // h_new
    #pragma unroll 8
    for (int k = 0; k < 512; ++k) a2 += s_v[k] * w[256 + k];        // context
    out1_ws[b * HH + j] = a2;
  }
}

// ---------------------------------------------------------------------------
// K5: logits = out1(32x256) @ W_out2.T(256x50000) + b_out2 via WMMA f16->f32.
// A (out1) is staged once per block into LDS as f16 and shared by all 8
// waves (ds_load_b128 frags); B (W_out2, the real 51 MB stream) comes from
// global with f32->f16 convert + WGP-scope prefetch. One wave per 16-wide
// vocab tile (3125 tiles exactly), 2 accumulators for M=32 sharing B.
// ---------------------------------------------------------------------------
__global__ __launch_bounds__(256) void k5_logits_wmma(
    const float* __restrict__ out1, const float* __restrict__ W_out2,
    const float* __restrict__ b_out2, float* __restrict__ logits)
{
  __shared__ _Float16 sA[BB * HH];   // 8192 halves = 16 KB
  const int t = threadIdx.x;
  // stage A: 8192 f32 -> f16, coalesced float4 loads
  #pragma unroll
  for (int i = 0; i < 8; ++i) {
    const int f4 = t + 256 * i;                 // 2048 float4 total
    const float4 g = ((const float4*)out1)[f4];
    const int e = f4 * 4;
    sA[e + 0] = (_Float16)g.x; sA[e + 1] = (_Float16)g.y;
    sA[e + 2] = (_Float16)g.z; sA[e + 3] = (_Float16)g.w;
  }
  __syncthreads();

  const int wave = t >> 5, lane = t & 31;
  const int tile = blockIdx.x * 8 + wave;          // wave-uniform
  if (tile >= VV / 16) return;                     // whole-wave exit after barrier
  const int nbase = tile * 16;
  const int half = lane >> 4, l16 = lane & 15;

  v8f acc0 = {}; v8f acc1 = {};
  const _Float16* a0p = sA + l16 * HH;             // M-tile 0: rows 0..15
  const _Float16* a1p = sA + (l16 + 16) * HH;      // M-tile 1: rows 16..31
  const float* brow = W_out2 + (size_t)(nbase + l16) * HH;

  #pragma unroll
  for (int kb = 0; kb < HH; kb += 32) {
    if (kb + 32 < HH) __builtin_prefetch(brow + kb + 32, 0, 3);
    const int base = kb + 8 * half;
    // A frags from LDS (16B-aligned ds_load_b128)
    const v8h a0lo = *(const v8h*)(a0p + base);
    const v8h a0hi = *(const v8h*)(a0p + base + 16);
    const v8h a1lo = *(const v8h*)(a1p + base);
    const v8h a1hi = *(const v8h*)(a1p + base + 16);
    const v16h a0 = __builtin_shufflevector(a0lo, a0hi, 0, 1, 2, 3, 4, 5, 6, 7,
                                            8, 9, 10, 11, 12, 13, 14, 15);
    const v16h a1 = __builtin_shufflevector(a1lo, a1hi, 0, 1, 2, 3, 4, 5, 6, 7,
                                            8, 9, 10, 11, 12, 13, 14, 15);
    // B frag from global, convert f32->f16
    const float4 g0 = ((const float4*)(brow + base))[0];
    const float4 g1 = ((const float4*)(brow + base))[1];
    const float4 g2 = ((const float4*)(brow + base + 16))[0];
    const float4 g3 = ((const float4*)(brow + base + 16))[1];
    v16h bf;
    bf[0]  = (_Float16)g0.x; bf[1]  = (_Float16)g0.y; bf[2]  = (_Float16)g0.z; bf[3]  = (_Float16)g0.w;
    bf[4]  = (_Float16)g1.x; bf[5]  = (_Float16)g1.y; bf[6]  = (_Float16)g1.z; bf[7]  = (_Float16)g1.w;
    bf[8]  = (_Float16)g2.x; bf[9]  = (_Float16)g2.y; bf[10] = (_Float16)g2.z; bf[11] = (_Float16)g2.w;
    bf[12] = (_Float16)g3.x; bf[13] = (_Float16)g3.y; bf[14] = (_Float16)g3.z; bf[15] = (_Float16)g3.w;

    acc0 = __builtin_amdgcn_wmma_f32_16x16x32_f16(false, a0, false, bf,
                                                  (short)0, acc0, false, false);
    acc1 = __builtin_amdgcn_wmma_f32_16x16x32_f16(false, a1, false, bf,
                                                  (short)0, acc1, false, false);
  }

  const float bias = b_out2[nbase + l16];
  #pragma unroll
  for (int r = 0; r < 8; ++r) {
    const int m = r + 8 * half;                    // C/D layout: lane n=l16, m=r+8*half
    logits[(size_t)m * VV + nbase + l16]        = acc0[r] + bias;
    logits[(size_t)(m + 16) * VV + nbase + l16] = acc1[r] + bias;
  }
}

// ---------------------------------------------------------------------------
// K6a: partial online-softmax stats over V: grid (B, NST) for concurrency
// ---------------------------------------------------------------------------
__global__ __launch_bounds__(256) void k6a_rowstats_partial(
    const float* __restrict__ logits, float* __restrict__ pmax,
    float* __restrict__ psum)
{
  __shared__ float sm[256], ss[256];
  const int b = blockIdx.x, chunk = blockIdx.y, t = threadIdx.x;
  const int v0 = chunk * (VV / NST);               // 6250 per chunk
  float m = -INFINITY, s = 0.f;
  for (int v = t; v < VV / NST; v += 256) {
    const float x = logits[(size_t)b * VV + v0 + v];
    if (x > m) { s = s * __expf(m - x) + 1.f; m = x; }
    else s += __expf(x - m);
  }
  sm[t] = m; ss[t] = s; __syncthreads();
  for (int off = 128; off > 0; off >>= 1) {
    if (t < off) {
      const float m2 = sm[t + off], s2 = ss[t + off];
      const float M = fmaxf(sm[t], m2);
      ss[t] = ss[t] * __expf(sm[t] - M) + s2 * __expf(m2 - M);
      sm[t] = M;
    }
    __syncthreads();
  }
  if (t == 0) { pmax[b * NST + chunk] = sm[0]; psum[b * NST + chunk] = ss[0]; }
}

// K6b: deterministic combine, one thread per batch row
__global__ __launch_bounds__(32) void k6b_rowstats_combine(
    const float* __restrict__ pmax, const float* __restrict__ psum,
    const float* __restrict__ pgen, float* __restrict__ rowmax,
    float* __restrict__ rowscale)
{
  const int b = threadIdx.x;
  float m = pmax[b * NST], s = psum[b * NST];
  #pragma unroll
  for (int c = 1; c < NST; ++c) {
    const float m2 = pmax[b * NST + c], s2 = psum[b * NST + c];
    const float M = fmaxf(m, m2);
    s = s * __expf(m - M) + s2 * __expf(m2 - M);
    m = M;
  }
  rowmax[b] = m;
  rowscale[b] = pgen[b] / s;
}

// ---------------------------------------------------------------------------
// K7: final_dist vocab part (p_gen * softmax; OOV tail = extra_zeros = 0)
// ---------------------------------------------------------------------------
__global__ __launch_bounds__(256) void k7_vocab_dist(
    const float* __restrict__ logits, const float* __restrict__ rowmax,
    const float* __restrict__ rowscale, float* __restrict__ finald)
{
  const size_t i = (size_t)blockIdx.x * 256 + threadIdx.x;
  if (i >= (size_t)BB * VE) return;
  const int b = (int)(i / VE);
  const int v = (int)(i % VE);
  float r = 0.f;
  if (v < VV) r = rowscale[b] * __expf(logits[(size_t)b * VV + v] - rowmax[b]);
  finald[i] = r;
}

// ---------------------------------------------------------------------------
// K8: scatter-add (1-p_gen)*attn at extended-vocab indices
// ---------------------------------------------------------------------------
__global__ __launch_bounds__(256) void k8_scatter(
    const float* __restrict__ attn, const float* __restrict__ pgen,
    const int* __restrict__ ext_vocab, float* __restrict__ finald)
{
  const int i = blockIdx.x * 256 + threadIdx.x;   // B*S = 65536 exactly
  const int b = i >> 11;
  const float w = (1.f - pgen[b]) * attn[i];
  atomicAdd(finald + (size_t)b * VE + ext_vocab[i], w);
}

// ---------------------------------------------------------------------------
extern "C" void kernel_launch(void* const* d_in, const int* in_sizes, int n_in,
                              void* d_out, int out_size, void* d_ws, size_t ws_size,
                              hipStream_t stream)
{
  const int*   inp        = (const int*)  d_in[0];
  const float* h0         = (const float*)d_in[1];
  const float* c0         = (const float*)d_in[2];
  const float* enc_out    = (const float*)d_in[3];
  const float* enc_feat   = (const float*)d_in[4];
  const float* enc_mask   = (const float*)d_in[6];
  const float* prev_ctx   = (const float*)d_in[8];
  const int*   ext_vocab  = (const int*)  d_in[10];
  const float* coverage   = (const float*)d_in[11];
  const float* emb        = (const float*)d_in[12];
  const float* W_c        = (const float*)d_in[13];
  const float* W_dp       = (const float*)d_in[14];
  const float* b_dp       = (const float*)d_in[15];
  const float* v_attn     = (const float*)d_in[16];
  const float* W_comb     = (const float*)d_in[21];
  const float* b_comb     = (const float*)d_in[22];
  const float* W_ih       = (const float*)d_in[23];
  const float* W_hh       = (const float*)d_in[24];
  const float* b_ih       = (const float*)d_in[25];
  const float* b_hh       = (const float*)d_in[26];
  const float* W_pgen     = (const float*)d_in[27];
  const float* b_pgen     = (const float*)d_in[28];
  const float* W_out1     = (const float*)d_in[29];
  const float* b_out1     = (const float*)d_in[30];
  const float* W_out2     = (const float*)d_in[31];
  const float* b_out2     = (const float*)d_in[32];
  // d_in[5,7,9,17..20,33]: section attention / extra_zeros / step — dead code

  float* out = (float*)d_out;
  float* ws  = (float*)d_ws;
  float* x_comb   = ws + WS_X;
  float* dec_feat = ws + WS_DF;
  float* scores   = ws + WS_SC;
  float* out1     = ws + WS_O1;
  float* rowmax   = ws + WS_MAX;
  float* rowscale = ws + WS_SCL;
  float* pmax     = ws + WS_PM;
  float* psum     = ws + WS_PS;
  float* ctxp     = ws + WS_CTXP;
  float* logits   = ws + WS_LOG;

  k1_embed_lstm<<<BB, 256, 0, stream>>>(
      inp, h0, c0, prev_ctx, emb, W_comb, b_comb, W_ih, W_hh, b_ih, b_hh,
      W_dp, b_dp, x_comb, dec_feat, out + O_H, out + O_C);

  k2_scores<<<dim3(BB, SS / 64), 256, 0, stream>>>(
      enc_feat, dec_feat, coverage, W_c, v_attn, scores);

  k3a_softmax<<<BB, 512, 0, stream>>>(
      scores, enc_mask, coverage, out + O_ATTN, out + O_COV);

  k3b_ctx_partial<<<dim3(BB, NCTX), 512, 0, stream>>>(
      out + O_ATTN, enc_out, ctxp);

  k3c_ctx_combine<<<BB, 512, 0, stream>>>(ctxp, out + O_CTX);

  k4_pgen_out1<<<BB, 128, 0, stream>>>(
      out + O_CTX, out + O_H, out + O_C, x_comb,
      W_pgen, b_pgen, W_out1, b_out1, out + O_PGEN, out1);

  k5_logits_wmma<<<(VV / 16 + 7) / 8, 256, 0, stream>>>(
      out1, W_out2, b_out2, logits);

  k6a_rowstats_partial<<<dim3(BB, NST), 256, 0, stream>>>(logits, pmax, psum);

  k6b_rowstats_combine<<<1, 32, 0, stream>>>(
      pmax, psum, out + O_PGEN, rowmax, rowscale);

  k7_vocab_dist<<<(int)(((size_t)BB * VE + 255) / 256), 256, 0, stream>>>(
      logits, rowmax, rowscale, out + O_FINAL);

  k8_scatter<<<(BB * SS) / 256, 256, 0, stream>>>(
      out + O_ATTN, out + O_PGEN, ext_vocab, out + O_FINAL);
}
